// GNN_51573967290999
// MI455X (gfx1250) — matrix operand order repaired and checked by reading, not compiled, not run on hardware
//
#include <hip/hip_runtime.h>
#include <hip/hip_bf16.h>
#include <math.h>

// ---------------- problem constants (match reference) ----------------
constexpr int cN0 = 400000, cN1 = 100000, cN2 = 20000;
constexpr int cE1 = 1000000, cE2 = 200000;
constexpr int cIN = 128, cHID = 256, cHEADS = 4, cOUT = 47;
constexpr int cC = cHID / cHEADS;           // 64
constexpr float cEPS = 1e-5f;
constexpr float cNEG = 0.2f;

typedef __attribute__((ext_vector_type(16))) __bf16 v16bf;
typedef __attribute__((ext_vector_type(8)))  __bf16 v8bf;
typedef __attribute__((ext_vector_type(4)))  __bf16 v4bf;
typedef __attribute__((ext_vector_type(4)))  float  v4f;
typedef __attribute__((ext_vector_type(8)))  float  v8f;

// ---------------- WMMA GEMM: Y[M,N] = A[M,K] @ Bt[N,K]^T (+bias) -----
// A, Bt pre-converted bf16; f32 accumulate. One wave -> 16 x (16*NT).
// 256 threads = 8 waves / block. K % 32 == 0, M % 16 == 0 assumed.
template<int NT>
__global__ __launch_bounds__(256) void wmma_gemm(
    const __bf16* __restrict__ A, const __bf16* __restrict__ Bt,
    const float* __restrict__ bias, float* __restrict__ Y,
    int M, int K, int N, int waveTilesN)
{
    const int wave = blockIdx.x * 8 + (threadIdx.x >> 5);
    const int lane = threadIdx.x & 31;
    const int tm = wave / waveTilesN;
    const int tn = wave % waveTilesN;
    const int m0 = tm * 16;
    if (m0 >= M) return;
    const int n0 = tn * 16 * NT;

    const int half = lane >> 4;      // K-half selector
    const int l16  = lane & 15;      // M (for A) / N (for B,C,D)

    v8f acc[NT];
    #pragma unroll
    for (int t = 0; t < NT; ++t) acc[t] = v8f{};

    int col[NT];
    const __bf16* __restrict__ Brow[NT];
    #pragma unroll
    for (int t = 0; t < NT; ++t) {
        col[t] = n0 + t * 16 + l16;
        int cc = col[t] < N ? col[t] : 0;   // clamp: garbage cols never stored
        Brow[t] = Bt + (size_t)cc * K + half * 8;
    }
    const __bf16* __restrict__ Arow = A + (size_t)(m0 + l16) * K + half * 8;

    for (int kk = 0; kk < K; kk += 32) {
        // A fragment: e 0..7 -> K = kk+half*8+e ; e 8..15 -> +16
        v8bf al = *(const v8bf*)(Arow + kk);
        v8bf ah = *(const v8bf*)(Arow + kk + 16);
        v16bf a;
        #pragma unroll
        for (int e = 0; e < 8; ++e) { a[e] = al[e]; a[e + 8] = ah[e]; }
        #pragma unroll
        for (int t = 0; t < NT; ++t) {
            // B fragment: e 0..7 -> K = kk+half*8+e ; e 8..15 -> +16
            v8bf bl = *(const v8bf*)(Brow[t] + kk);
            v8bf bh = *(const v8bf*)(Brow[t] + kk + 16);
            v16bf b;
            #pragma unroll
            for (int e = 0; e < 8; ++e) { b[e] = bl[e]; b[e + 8] = bh[e]; }
            acc[t] = __builtin_amdgcn_wmma_f32_16x16x32_bf16(
                false, a, false, b, (short)0, acc[t], false, false);
        }
    }
    #pragma unroll
    for (int t = 0; t < NT; ++t) {
        if (col[t] >= N) continue;
        float bv = bias ? bias[col[t]] : 0.0f;
        #pragma unroll
        for (int r = 0; r < 8; ++r) {
            int row = m0 + half * 8 + r;
            Y[(size_t)row * N + col[t]] = acc[t][r] + bv;
        }
    }
}

// ------- fp32 [M,K] -> bf16 [M,K], 4 elements / thread, vectorized -------
__global__ void convert_bf16(const float* __restrict__ in, __bf16* __restrict__ out,
                             long long n4)
{
    long long i = (long long)blockIdx.x * blockDim.x + threadIdx.x;
    if (i >= n4) return;
    v4f v = *(const v4f*)(in + i * 4);
    v4bf o;
    #pragma unroll
    for (int e = 0; e < 4; ++e) o[e] = (__bf16)v[e];
    *(v4bf*)(out + i * 4) = o;
}

// ------- fp32 W[K,N] -> bf16 Wt[N,K], LDS-tiled transpose ----------------
__global__ __launch_bounds__(256) void transpose_bf16(
    const float* __restrict__ W, __bf16* __restrict__ Wt, int K, int N)
{
    __shared__ float tile[32][33];
    int k0 = blockIdx.x * 32, n0 = blockIdx.y * 32;
    int tx = threadIdx.x, ty = threadIdx.y;          // block (32,8)
    #pragma unroll
    for (int j = 0; j < 4; ++j) {
        int k = k0 + ty + j * 8, n = n0 + tx;
        tile[ty + j * 8][tx] = (k < K && n < N) ? W[(size_t)k * N + n] : 0.0f;
    }
    __syncthreads();
    #pragma unroll
    for (int j = 0; j < 4; ++j) {
        int n = n0 + ty + j * 8, k = k0 + tx;
        if (n < N && k < K) Wt[(size_t)n * K + k] = (__bf16)tile[tx][ty + j * 8];
    }
}

// ---------------- attention logits: out[i,h] = <xp[i,h*C:..], a[h,:]> ----
__global__ void att_logits(const float* __restrict__ xp, const float* __restrict__ avec,
                           float* __restrict__ out, int n)
{
    int idx = blockIdx.x * blockDim.x + threadIdx.x;
    if (idx >= n * cHEADS) return;
    int i = idx >> 2, h = idx & 3;
    const float* xr = xp + (size_t)i * cHID + h * cC;
    const float* ar = avec + h * cC;
    float s = 0.f;
    #pragma unroll 8
    for (int c = 0; c < cC; ++c) s += xr[c] * ar[c];
    out[idx] = s;
}

__global__ void fill_neg_inf(float* p, int n)
{
    int i = blockIdx.x * blockDim.x + threadIdx.x;
    if (i < n) p[i] = -__builtin_inff();
}

__device__ inline void atomicMaxF(float* addr, float value)
{
    if (value >= 0.f) atomicMax((int*)addr, __float_as_int(value));
    else              atomicMin((unsigned int*)addr, __float_as_uint(value));
}

__global__ void edge_max(const float* __restrict__ als, const float* __restrict__ ald,
                         const int* __restrict__ src, const int* __restrict__ dst,
                         float* __restrict__ m, int E)
{
    int idx = blockIdx.x * blockDim.x + threadIdx.x;
    if (idx >= E * cHEADS) return;
    int e = idx >> 2, h = idx & 3;
    float v = als[src[e] * cHEADS + h] + ald[dst[e] * cHEADS + h];
    v = v > 0.f ? v : cNEG * v;
    atomicMaxF(&m[dst[e] * cHEADS + h], v);
}

__global__ void edge_exp(const float* __restrict__ als, const float* __restrict__ ald,
                         const int* __restrict__ src, const int* __restrict__ dst,
                         const float* __restrict__ m, float* __restrict__ ee,
                         float* __restrict__ s, int E)
{
    int idx = blockIdx.x * blockDim.x + threadIdx.x;
    if (idx >= E * cHEADS) return;
    int e = idx >> 2, h = idx & 3;
    int de = dst[e];
    float v = als[src[e] * cHEADS + h] + ald[de * cHEADS + h];
    v = v > 0.f ? v : cNEG * v;
    float ex = expf(v - m[de * cHEADS + h]);
    ee[idx] = ex;
    atomicAdd(&s[de * cHEADS + h], ex);
}

// one wave per edge; 8 channels per lane
__global__ __launch_bounds__(256) void edge_aggregate(
    const float* __restrict__ xp, const float* __restrict__ ee, const float* __restrict__ s,
    const int* __restrict__ src, const int* __restrict__ dst,
    float* __restrict__ agg, int E)
{
    int e = blockIdx.x * 8 + (threadIdx.x >> 5);
    if (e >= E) return;
    int lane = threadIdx.x & 31;
    int se = src[e], de = dst[e];
    float alpha[cHEADS];
    #pragma unroll
    for (int h = 0; h < cHEADS; ++h)
        alpha[h] = ee[(size_t)e * cHEADS + h] / (s[(size_t)de * cHEADS + h] + 1e-16f);
    const float* xr = xp + (size_t)se * cHID;
    float* orow = agg + (size_t)de * cHID;
    #pragma unroll
    for (int j = 0; j < cHID / 32; ++j) {
        int c = lane + j * 32;
        atomicAdd(&orow[c], xr[c] * alpha[c >> 6]);
    }
}

// pre[i,c] = skip[i,c](in place) + agg[i,c] + b[c]
__global__ void combine_add(float* __restrict__ pre, const float* __restrict__ agg,
                            const float* __restrict__ b, int n)
{
    size_t i = (size_t)blockIdx.x * blockDim.x + threadIdx.x;
    if (i >= (size_t)n * cHID) return;
    pre[i] += agg[i] + b[i & (cHID - 1)];
}

__global__ __launch_bounds__(256) void bn_stats(const float* __restrict__ x,
                                                float* __restrict__ sums, int n)
{
    int c = threadIdx.x;           // channel (256 = HID)
    int r0 = blockIdx.x * 256;
    int r1 = r0 + 256; if (r1 > n) r1 = n;
    float s = 0.f, s2 = 0.f;
    for (int r = r0; r < r1; ++r) {
        float v = x[(size_t)r * cHID + c];
        s += v; s2 += v * v;
    }
    atomicAdd(&sums[c], s);
    atomicAdd(&sums[cHID + c], s2);
}

__global__ void bn_finalize(const float* __restrict__ sums, const float* __restrict__ g,
                            const float* __restrict__ be, float* __restrict__ ss, float inv_n)
{
    int c = threadIdx.x;
    float mu  = sums[c] * inv_n;
    float var = sums[cHID + c] * inv_n - mu * mu;
    float sc  = g[c] * rsqrtf(var + cEPS);
    ss[c] = sc;
    ss[cHID + c] = be[c] - mu * sc;
}

// BN + activation, emitting bf16 (consumed only by subsequent GEMMs).
// mode 0 = ELU, 1 = ReLU.
__global__ void bn_apply_emit(const float* __restrict__ x, const float* __restrict__ ss,
                              __bf16* __restrict__ out, int n, int mode)
{
    size_t i = (size_t)blockIdx.x * blockDim.x + threadIdx.x;
    if (i >= (size_t)n * cHID) return;
    int c = (int)(i & (cHID - 1));
    float v = x[i] * ss[c] + ss[cHID + c];
    if (mode == 0) v = v > 0.f ? v : expm1f(v);
    else           v = v > 0.f ? v : 0.f;
    out[i] = (__bf16)v;
}

// ---------------------------------------------------------------------
extern "C" void kernel_launch(void* const* d_in, const int* in_sizes, int n_in,
                              void* d_out, int out_size, void* d_ws, size_t ws_size,
                              hipStream_t stream)
{
    const float* x     = (const float*)d_in[0];
    const int*   src1  = (const int*)d_in[1];
    const int*   dst1  = (const int*)d_in[2];
    const int*   src2  = (const int*)d_in[3];
    const int*   dst2  = (const int*)d_in[4];
    const float* W1    = (const float*)d_in[5];
    const float* a_s1  = (const float*)d_in[6];
    const float* a_d1  = (const float*)d_in[7];
    const float* b1    = (const float*)d_in[8];
    const float* Wsk1  = (const float*)d_in[9];
    const float* bsk1  = (const float*)d_in[10];
    const float* g1    = (const float*)d_in[11];
    const float* be1   = (const float*)d_in[12];
    const float* W2    = (const float*)d_in[13];
    const float* a_s2  = (const float*)d_in[14];
    const float* a_d2  = (const float*)d_in[15];
    const float* b2    = (const float*)d_in[16];
    const float* Wsk2  = (const float*)d_in[17];
    const float* bsk2  = (const float*)d_in[18];
    const float* g2    = (const float*)d_in[19];
    const float* be2   = (const float*)d_in[20];
    const float* Wm1   = (const float*)d_in[21];
    const float* bm1   = (const float*)d_in[22];
    const float* gm    = (const float*)d_in[23];
    const float* bem   = (const float*)d_in[24];
    const float* Wm2   = (const float*)d_in[25];
    const float* bm2   = (const float*)d_in[26];
    float* out = (float*)d_out;
    (void)in_sizes; (void)n_in; (void)out_size; (void)ws_size;

    // ---- workspace layout (byte allocator, 256B aligned) ----
    char* wsb = (char*)d_ws;
    size_t off = 0;
    auto allocB = [&](size_t bytes) -> void* {
        void* p = wsb + off;
        off += (bytes + 255) & ~(size_t)255;
        return p;
    };
    auto allocF  = [&](size_t nf) -> float*  { return (float*) allocB(nf * 4); };
    auto allocH  = [&](size_t nf) -> __bf16* { return (__bf16*)allocB(nf * 2); };

    float*  xp1  = allocF((size_t)cN0 * cHID);
    float*  als1 = allocF((size_t)cN0 * cHEADS);
    float*  ald1 = allocF((size_t)cN1 * cHEADS);
    float*  m1   = allocF((size_t)cN1 * cHEADS);
    float*  s1v  = allocF((size_t)cN1 * cHEADS);
    float*  ee1  = allocF((size_t)cE1 * cHEADS);
    float*  agg1 = allocF((size_t)cN1 * cHID);
    float*  pre1 = allocF((size_t)cN1 * cHID);     // skip -> pre (in place)
    float*  xp2  = allocF((size_t)cN1 * cHID);
    float*  als2 = allocF((size_t)cN1 * cHEADS);
    float*  ald2 = allocF((size_t)cN2 * cHEADS);
    float*  m2   = allocF((size_t)cN2 * cHEADS);
    float*  s2v  = allocF((size_t)cN2 * cHEADS);
    float*  ee2  = allocF((size_t)cE2 * cHEADS);
    float*  agg2 = allocF((size_t)cN2 * cHID);
    float*  pre2 = allocF((size_t)cN2 * cHID);
    float*  mh   = allocF((size_t)cN2 * cHID);
    float*  sumsA = allocF(2 * cHID); float* ssA = allocF(2 * cHID);
    float*  sumsB = allocF(2 * cHID); float* ssB = allocF(2 * cHID);
    float*  sumsC = allocF(2 * cHID); float* ssC = allocF(2 * cHID);
    // bf16 operand buffers
    __bf16* xb    = allocH((size_t)cN0 * cIN);
    __bf16* h1b   = allocH((size_t)cN1 * cHID);
    __bf16* h2b   = allocH((size_t)cN2 * cHID);
    __bf16* mhb   = allocH((size_t)cN2 * cHID);
    __bf16* Wt1   = allocH((size_t)cHID * cIN);
    __bf16* Wskt1 = allocH((size_t)cHID * cIN);
    __bf16* Wt2   = allocH((size_t)cHID * cHID);
    __bf16* Wskt2 = allocH((size_t)cHID * cHID);
    __bf16* Wtm1  = allocH((size_t)cHID * cHID);
    __bf16* Wtm2  = allocH((size_t)cOUT * cHID);

    auto gemm4 = [&](const __bf16* A, const __bf16* Bt, const float* bias, float* Y,
                     int M, int K, int N) {
        int wtn = N / 64;
        int waves = ((M + 15) / 16) * wtn;
        wmma_gemm<4><<<(waves + 7) / 8, 256, 0, stream>>>(A, Bt, bias, Y, M, K, N, wtn);
    };
    auto gemm1 = [&](const __bf16* A, const __bf16* Bt, const float* bias, float* Y,
                     int M, int K, int N) {
        int wtn = (N + 15) / 16;
        int waves = ((M + 15) / 16) * wtn;
        wmma_gemm<1><<<(waves + 7) / 8, 256, 0, stream>>>(A, Bt, bias, Y, M, K, N, wtn);
    };
    auto trans = [&](const float* W, __bf16* Wt, int K, int N) {
        transpose_bf16<<<dim3((K + 31) / 32, (N + 31) / 32), dim3(32, 8), 0, stream>>>(W, Wt, K, N);
    };
    auto conv = [&](const float* in, __bf16* o, long long n) {
        long long n4 = n / 4;
        convert_bf16<<<(unsigned)((n4 + 255) / 256), 256, 0, stream>>>(in, o, n4);
    };
    auto grid1d = [](long long n) { return (unsigned)((n + 255) / 256); };

    // ---- prepare bf16 operands ----
    trans(W1,   Wt1,   cIN,  cHID);
    trans(Wsk1, Wskt1, cIN,  cHID);
    trans(W2,   Wt2,   cHID, cHID);
    trans(Wsk2, Wskt2, cHID, cHID);
    trans(Wm1,  Wtm1,  cHID, cHID);
    trans(Wm2,  Wtm2,  cHID, cOUT);
    conv(x, xb, (long long)cN0 * cIN);

    // ================= Layer 1 =================
    hipMemsetAsync(agg1, 0, (size_t)cN1 * cHID * sizeof(float), stream);
    hipMemsetAsync(s1v, 0, (size_t)cN1 * cHEADS * sizeof(float), stream);
    fill_neg_inf<<<grid1d((long long)cN1 * cHEADS), 256, 0, stream>>>(m1, cN1 * cHEADS);

    gemm4(xb, Wt1, nullptr, xp1, cN0, cIN, cHID);
    att_logits<<<grid1d((long long)cN0 * cHEADS), 256, 0, stream>>>(xp1, a_s1, als1, cN0);
    att_logits<<<grid1d((long long)cN1 * cHEADS), 256, 0, stream>>>(xp1, a_d1, ald1, cN1);
    edge_max<<<grid1d((long long)cE1 * cHEADS), 256, 0, stream>>>(als1, ald1, src1, dst1, m1, cE1);
    edge_exp<<<grid1d((long long)cE1 * cHEADS), 256, 0, stream>>>(als1, ald1, src1, dst1, m1, ee1, s1v, cE1);
    gemm4(xb, Wskt1, bsk1, pre1, cN1, cIN, cHID);
    edge_aggregate<<<(cE1 + 7) / 8, 256, 0, stream>>>(xp1, ee1, s1v, src1, dst1, agg1, cE1);
    combine_add<<<grid1d((long long)cN1 * cHID), 256, 0, stream>>>(pre1, agg1, b1, cN1);
    hipMemsetAsync(sumsA, 0, 2 * cHID * sizeof(float), stream);
    bn_stats<<<(cN1 + 255) / 256, 256, 0, stream>>>(pre1, sumsA, cN1);
    bn_finalize<<<1, cHID, 0, stream>>>(sumsA, g1, be1, ssA, 1.0f / cN1);
    bn_apply_emit<<<grid1d((long long)cN1 * cHID), 256, 0, stream>>>(pre1, ssA, h1b, cN1, 0);

    // ================= Layer 2 =================
    hipMemsetAsync(agg2, 0, (size_t)cN2 * cHID * sizeof(float), stream);
    hipMemsetAsync(s2v, 0, (size_t)cN2 * cHEADS * sizeof(float), stream);
    fill_neg_inf<<<grid1d((long long)cN2 * cHEADS), 256, 0, stream>>>(m2, cN2 * cHEADS);

    gemm4(h1b, Wt2, nullptr, xp2, cN1, cHID, cHID);
    att_logits<<<grid1d((long long)cN1 * cHEADS), 256, 0, stream>>>(xp2, a_s2, als2, cN1);
    att_logits<<<grid1d((long long)cN2 * cHEADS), 256, 0, stream>>>(xp2, a_d2, ald2, cN2);
    edge_max<<<grid1d((long long)cE2 * cHEADS), 256, 0, stream>>>(als2, ald2, src2, dst2, m2, cE2);
    edge_exp<<<grid1d((long long)cE2 * cHEADS), 256, 0, stream>>>(als2, ald2, src2, dst2, m2, ee2, s2v, cE2);
    gemm4(h1b, Wskt2, bsk2, pre2, cN2, cHID, cHID);
    edge_aggregate<<<(cE2 + 7) / 8, 256, 0, stream>>>(xp2, ee2, s2v, src2, dst2, agg2, cE2);
    combine_add<<<grid1d((long long)cN2 * cHID), 256, 0, stream>>>(pre2, agg2, b2, cN2);
    hipMemsetAsync(sumsB, 0, 2 * cHID * sizeof(float), stream);
    bn_stats<<<(cN2 + 255) / 256, 256, 0, stream>>>(pre2, sumsB, cN2);
    bn_finalize<<<1, cHID, 0, stream>>>(sumsB, g2, be2, ssB, 1.0f / cN2);
    bn_apply_emit<<<grid1d((long long)cN2 * cHID), 256, 0, stream>>>(pre2, ssB, h2b, cN2, 0);

    // ================= MLP head =================
    gemm4(h2b, Wtm1, bm1, mh, cN2, cHID, cHID);
    hipMemsetAsync(sumsC, 0, 2 * cHID * sizeof(float), stream);
    bn_stats<<<(cN2 + 255) / 256, 256, 0, stream>>>(mh, sumsC, cN2);
    bn_finalize<<<1, cHID, 0, stream>>>(sumsC, gm, bem, ssC, 1.0f / cN2);
    bn_apply_emit<<<grid1d((long long)cN2 * cHID), 256, 0, stream>>>(mh, ssC, mhb, cN2, 1);
    gemm1(mhb, Wtm2, bm2, out, cN2, cHID, cOUT);
}
